// Prediction_rep_44461501448284
// MI455X (gfx1250) — matrix operand order repaired and checked by reading, not compiled, not run on hardware
//
#include <hip/hip_runtime.h>
#include <hip/hip_bf16.h>

typedef __attribute__((ext_vector_type(16))) _Float16 v16h;
typedef __attribute__((ext_vector_type(8)))  float    v8f;

#define OBS_D 2064
#define STEPC 344

// ---- workspace f16 weight offsets (elements), all K padded to mult of 32 ----
enum : int {
  OFF_WSE = 0,                     // 32 x 64  (K=15)
  OFF_WDE = OFF_WSE + 32 * 64,     // 64 x 32
  OFF_WH1 = OFF_WDE + 64 * 32,     // 32 x 64
  OFF_WH2 = OFF_WH1 + 32 * 64,     // 32 x 64
  OFF_WHO = OFF_WH2 + 32 * 64,     // 128 x 64
  OFF_WC1 = OFF_WHO + 128 * 64,    // 32 x 16  (K=6)
  OFF_WC2 = OFF_WC1 + 32 * 16,     // 32 x 16
  OFF_WCO = OFF_WC2 + 32 * 16,     // 32 x 64
  OFF_WL1 = OFF_WCO + 32 * 64,     // 128 x 64 (K=108)
  OFF_WL2 = OFF_WL1 + 128 * 64,    // 64 x 64
  OFF_WM1 = OFF_WL2 + 64 * 64,     // 224 x 64 (K=205)
  OFF_WM2 = OFF_WM1 + 224 * 64,    // 64 x 64
};

// ---- dynamic LDS layout (f16 elements) ----
enum : int {
  L_COMB = 0,                      // 16 x 224 final-MLP input tile
  L_X    = L_COMB + 16 * 224,      // 16 x 128 generic input staging
  L_HID  = L_X    + 16 * 128,      // 16 x 64  MLP hidden staging
  L_EMB  = L_HID  + 16 * 64,       // 13 x 16 x 32 node embeddings / cav feats
  L_WH   = L_EMB  + 13 * 16 * 32,  // 13 x 16 x 64 current-layer Wh
  L_GAT  = L_WH   + 13 * 16 * 64,  // 13 x 16 x 128 GAT concat buffer
  L_F16_TOT = L_GAT + 13 * 16 * 128,
};
// f32 tail: Si(13*16) + Sj(13*16) + async-staged cur slice (16 x 192)
#define SMEM_BYTES ((size_t)(L_F16_TOT * 2 + (2 * 13 * 16 + 16 * 192) * 4))

// ---------------- WMMA helpers (wave32, 16x16x32 f16 -> f32) ----------------
__device__ __forceinline__ v8f wmma_f16(v16h a, v16h b, v8f c) {
  return __builtin_amdgcn_wmma_f32_16x16x32_f16(false, a, false, b, (short)0, c,
                                                false, false);
}

// A fragment: 16x32 f16 tile (row-major, row stride ld) starting at column k0.
__device__ __forceinline__ v16h ldA(const _Float16* base, int ld, int k0) {
  int l = threadIdx.x & 31;
  const _Float16* p = base + (l & 15) * ld + k0 + ((l >> 4) << 3);
  v16h a;
#pragma unroll
  for (int e = 0; e < 8; ++e) a[e] = p[e];
#pragma unroll
  for (int e = 8; e < 16; ++e) a[e] = p[e + 8];
  return a;
}

// B fragment: 32x16 slice of row-major [K][ldn] weight, rows k0.., cols n0..
__device__ __forceinline__ v16h ldB(const _Float16* W, int ldn, int k0, int n0) {
  int l = threadIdx.x & 31;
  const _Float16* p = W + (size_t)(k0 + ((l >> 4) << 3)) * ldn + n0 + (l & 15);
  v16h b;
#pragma unroll
  for (int e = 0; e < 16; ++e) b[e] = p[(size_t)(e < 8 ? e : e + 8) * ldn];
  return b;
}

// ---------------- GAT scalar helpers (star adjacency) ----------------
__device__ void compute_s(const _Float16* Wh, int nN, int F, const float* avec,
                          float* Si, float* Sj) {
  for (int p = threadIdx.x; p < nN * 16; p += 32) {
    int n = p >> 4, s = p & 15;
    const _Float16* w = Wh + (n * 16 + s) * F;
    float si = 0.f, sj = 0.f;
    for (int f = 0; f < F; ++f) {
      float v = (float)w[f];
      si += v * avec[f];
      sj += v * avec[F + f];
    }
    Si[n * 16 + s] = si;
    Sj[n * 16 + s] = sj;
  }
}

// att @ Wh with star adjacency: rows i>=1 -> Wh[0]; row 0 -> softmax over all j.
__device__ void att_star_apply(const _Float16* Wh, int nN, int F,
                               const float* Si, const float* Sj,
                               _Float16* outB, int outLd, int foff) {
  int tot = (nN - 1) * 16 * F;
  for (int idx = threadIdx.x; idx < tot; idx += 32) {
    int f = idx % F;
    int t = idx / F;
    int s = t & 15;
    int i = (t >> 4) + 1;
    outB[(i * 16 + s) * outLd + foff + f] = Wh[s * F + f];
  }
  for (int s = threadIdx.x; s < 16; s += 32) {
    float si0 = Si[s];
    float e[16];
    float m = -3.4e38f;
    for (int j = 0; j < nN; ++j) {
      float x = si0 + Sj[j * 16 + s];
      x = x > 0.f ? x : 0.2f * x;
      e[j] = x;
      if (x > m) m = x;
    }
    float sum = 0.f;
    for (int j = 0; j < nN; ++j) { e[j] = __expf(e[j] - m); sum += e[j]; }
    float inv = 1.f / sum;
    for (int f = 0; f < F; ++f) {
      float acc = 0.f;
      for (int j = 0; j < nN; ++j) acc += e[j] * (float)Wh[(j * 16 + s) * F + f];
      outB[s * outLd + foff + f] = (_Float16)(acc * inv);
    }
  }
}

// elu(att@Wh) then mean over nodes (star): (elu(row0) + (nN-1)*elu(Wh[0]))/nN
__device__ void att_star_mean_elu(const _Float16* Wh, int nN, int F,
                                  const float* Si, const float* Sj,
                                  _Float16* comb, int combOff) {
  for (int s = threadIdx.x; s < 16; s += 32) {
    float si0 = Si[s];
    float e[16];
    float m = -3.4e38f;
    for (int j = 0; j < nN; ++j) {
      float x = si0 + Sj[j * 16 + s];
      x = x > 0.f ? x : 0.2f * x;
      e[j] = x;
      if (x > m) m = x;
    }
    float sum = 0.f;
    for (int j = 0; j < nN; ++j) { e[j] = __expf(e[j] - m); sum += e[j]; }
    float inv = 1.f / sum;
    for (int f = 0; f < F; ++f) {
      float o0 = 0.f;
      for (int j = 0; j < nN; ++j) o0 += e[j] * (float)Wh[(j * 16 + s) * F + f];
      o0 *= inv;
      float w0 = (float)Wh[s * F + f];
      float e0 = o0 > 0.f ? o0 : __expf(o0) - 1.f;
      float ew = w0 > 0.f ? w0 : __expf(w0) - 1.f;
      comb[s * 224 + combOff + f] = (_Float16)((e0 + (float)(nN - 1) * ew) / (float)nN);
    }
  }
}

// ---------------- weight conversion: f32 -> f16 with K zero-pad ----------------
__global__ void cvt_pad_f16(const float* __restrict__ src, _Float16* __restrict__ dst,
                            int K, int Kpad, int N) {
  int i = blockIdx.x * blockDim.x + threadIdx.x;
  if (i >= Kpad * N) return;
  int k = i / N, n = i - k * N;
  dst[i] = (_Float16)((k < K) ? src[k * N + n] : 0.f);
}

// ---------------- fused prediction kernel: 1 wave handles 16 batch rows --------
__global__ __launch_bounds__(32) void pred_main(
    const float* __restrict__ obs, const float* b_se, const float* b_de,
    const float* ah1, const float* ah2, const float* aho, const float* ac1,
    const float* ac2, const float* aco, const float* bl1, const float* bl2,
    const float* bm1, const float* bm2, const _Float16* __restrict__ wf,
    float* __restrict__ out) {
  extern __shared__ char smem_raw[];
  _Float16* sf = (_Float16*)smem_raw;
  _Float16* shComb = sf + L_COMB;
  _Float16* shX = sf + L_X;
  _Float16* shHid = sf + L_HID;
  _Float16* shEmb = sf + L_EMB;
  _Float16* shWh = sf + L_WH;
  _Float16* shGat = sf + L_GAT;
  float* shSi = (float*)(sf + L_F16_TOT);
  float* shSj = shSi + 13 * 16;
  float* shCur = shSj + 13 * 16;  // 16 x 192: cur[64..255] per sample (f32)

  const int s0 = blockIdx.x * 16;
  const int lane = threadIdx.x & 31;
  const int cm = (lane >> 4) << 3;  // C-fragment base row
  const int cn = lane & 15;         // C-fragment col

  // ===== Async prefetch (CDNA5 ASYNCcnt path): cur cols 64..255 -> LDS =======
  // Issued up-front; the DMA overlaps stages 1-3 and is drained before stage 4.
  {
    unsigned ldsBase = (unsigned)(size_t)shCur;
    for (int i = threadIdx.x; i < 16 * 48; i += 32) {  // 48 b128 per sample
      int s = i / 48, q = i - s * 48;
      unsigned long long g =
          (unsigned long long)(obs + (size_t)(s0 + s) * OBS_D + 5 * STEPC + 64 + q * 4);
      unsigned lds = ldsBase + (unsigned)((s * 192 + q * 4) * 4);
      asm volatile("global_load_async_to_lds_b128 %0, %1, off"
                   :: "v"(lds), "v"(g) : "memory");
    }
  }

  // ===== Preload per-lane bias registers (uniform across all node tiles) =====
  float rb_se[4], rb_de[2], rb_l1[4], rb_l2[4], rb_m1[4], rb_m2[4];
#pragma unroll
  for (int j = 0; j < 4; ++j) {
    rb_se[j] = b_se[16 * j + cn];
    rb_l1[j] = bl1[16 * j + cn];
    rb_l2[j] = bl2[16 * j + cn];
    rb_m1[j] = bm1[16 * j + cn];
    rb_m2[j] = bm2[16 * j + cn];
  }
#pragma unroll
  for (int j = 0; j < 2; ++j) rb_de[j] = b_de[16 * j + cn];

  // ===== Stage 1: shared encoder (15 -> 64 lrelu0.1 -> 32) for 13 hdv nodes ===
  {
    v16h bWse[4], bWde[2][2];  // hoisted: reused across all 13 node tiles
#pragma unroll
    for (int j = 0; j < 4; ++j) bWse[j] = ldB(wf + OFF_WSE, 64, 0, 16 * j);
#pragma unroll
    for (int ks = 0; ks < 2; ++ks)
#pragma unroll
      for (int j = 0; j < 2; ++j)
        bWde[ks][j] = ldB(wf + OFF_WDE, 32, ks * 32, 16 * j);

    for (int n = 0; n < 13; ++n) {
      for (int idx = threadIdx.x; idx < 16 * 32; idx += 32) {
        int s = idx >> 5, k = idx & 31;
        float v = 0.f;
        if (k < 15) {
          int c = k / 3, d = k - c * 3;
          const float* row = obs + (size_t)(s0 + s) * OBS_D + (c + 1) * STEPC;
          v = (n == 0) ? row[234 + d] : row[(n - 1) * 6 + d];
        }
        shX[idx] = (_Float16)v;
      }
      __syncthreads();
      v16h a = ldA(shX, 32, 0);
      v8f zero = {};
#pragma unroll
      for (int j = 0; j < 4; ++j) {
        v8f c = wmma_f16(a, bWse[j], zero);
#pragma unroll
        for (int g = 0; g < 8; ++g) {
          float v = c[g] + rb_se[j];
          v = v > 0.f ? v : 0.1f * v;
          shHid[(cm + g) * 64 + 16 * j + cn] = (_Float16)v;
        }
      }
      __syncthreads();
      v8f acc2[2] = {};
#pragma unroll
      for (int ks = 0; ks < 2; ++ks) {
        v16h a2 = ldA(shHid, 64, ks * 32);
#pragma unroll
        for (int j = 0; j < 2; ++j) acc2[j] = wmma_f16(a2, bWde[ks][j], acc2[j]);
      }
#pragma unroll
      for (int j = 0; j < 2; ++j)
#pragma unroll
        for (int g = 0; g < 8; ++g)
          shEmb[(n * 16 + cm + g) * 32 + 16 * j + cn] =
              (_Float16)(acc2[j][g] + rb_de[j]);
      __syncthreads();
    }
  }

  // ===== Stage 2: hdv GAT layers 1,2 (32 -> 64), star attention ===============
  for (int layer = 0; layer < 2; ++layer) {
    const _Float16* Wl = wf + (layer == 0 ? OFF_WH1 : OFF_WH2);
    const float* av = (layer == 0) ? ah1 : ah2;
    v16h bW[4];
#pragma unroll
    for (int j = 0; j < 4; ++j) bW[j] = ldB(Wl, 64, 0, 16 * j);
    for (int n = 0; n < 13; ++n) {
      v16h a = ldA(shEmb + n * 16 * 32, 32, 0);
      v8f zero = {};
#pragma unroll
      for (int j = 0; j < 4; ++j) {
        v8f c = wmma_f16(a, bW[j], zero);
#pragma unroll
        for (int g = 0; g < 8; ++g)
          shWh[(n * 16 + cm + g) * 64 + 16 * j + cn] = (_Float16)c[g];
      }
    }
    __syncthreads();
    compute_s(shWh, 13, 64, av, shSi, shSj);
    __syncthreads();
    att_star_apply(shWh, 13, 64, shSi, shSj, shGat, 128, layer * 64);
    __syncthreads();
  }

  // ===== Stage 3: hdv GAT output layer (128 -> 64), elu + mean -> comb[0:64] ==
  for (int n = 0; n < 13; ++n) {
    v8f acc[4] = {};
#pragma unroll
    for (int ks = 0; ks < 4; ++ks) {
      v16h a = ldA(shGat + n * 16 * 128, 128, ks * 32);
#pragma unroll
      for (int j = 0; j < 4; ++j)
        acc[j] = wmma_f16(a, ldB(wf + OFF_WHO, 64, ks * 32, 16 * j), acc[j]);
    }
#pragma unroll
    for (int j = 0; j < 4; ++j)
#pragma unroll
      for (int g = 0; g < 8; ++g)
        shWh[(n * 16 + cm + g) * 64 + 16 * j + cn] = (_Float16)acc[j][g];
  }
  __syncthreads();
  compute_s(shWh, 13, 64, aho, shSi, shSj);
  __syncthreads();
  att_star_mean_elu(shWh, 13, 64, shSi, shSj, shComb, 0);
  __syncthreads();

  // ===== Drain the async prefetch; stages 4-6 read cur slice from LDS ========
  asm volatile("s_wait_asynccnt 0x0" ::: "memory");
  __syncthreads();

  // ===== Stage 4: cav GAT (9 nodes, 6 feats) =================================
  for (int idx = threadIdx.x; idx < 9 * 16 * 32; idx += 32) {
    int k = idx & 31;
    int s = (idx >> 5) & 15;
    int n = idx >> 9;
    float v = 0.f;
    if (k < 6) {
      const float* curS = shCur + s * 192;  // cur[64 + i]
      if (n == 0) {
        const int map0[6] = {170, 171, 172, 189, 187, 188};  // 234..236,253,251,252
        v = curS[map0[k]];
      } else {
        v = curS[8 + (n - 1) * 6 + k];  // cur[72 + ...]
      }
    }
    shEmb[idx] = (_Float16)v;
  }
  __syncthreads();
  for (int layer = 0; layer < 2; ++layer) {
    const _Float16* Wl = wf + (layer == 0 ? OFF_WC1 : OFF_WC2);
    const float* av = (layer == 0) ? ac1 : ac2;
    v16h bWc = ldB(Wl, 16, 0, 0);
    for (int n = 0; n < 9; ++n) {
      v16h a = ldA(shEmb + n * 16 * 32, 32, 0);
      v8f zero = {};
      v8f c = wmma_f16(a, bWc, zero);
#pragma unroll
      for (int g = 0; g < 8; ++g)
        shWh[(n * 16 + cm + g) * 16 + cn] = (_Float16)c[g];
    }
    __syncthreads();
    compute_s(shWh, 9, 16, av, shSi, shSj);
    __syncthreads();
    att_star_apply(shWh, 9, 16, shSi, shSj, shGat, 32, layer * 16);
    __syncthreads();
  }
  {
    v16h bWco[4];
#pragma unroll
    for (int j = 0; j < 4; ++j) bWco[j] = ldB(wf + OFF_WCO, 64, 0, 16 * j);
    for (int n = 0; n < 9; ++n) {
      v16h a = ldA(shGat + n * 16 * 32, 32, 0);
      v8f zero = {};
#pragma unroll
      for (int j = 0; j < 4; ++j) {
        v8f c = wmma_f16(a, bWco[j], zero);
#pragma unroll
        for (int g = 0; g < 8; ++g)
          shWh[(n * 16 + cm + g) * 64 + 16 * j + cn] = (_Float16)c[g];
      }
    }
  }
  __syncthreads();
  compute_s(shWh, 9, 64, aco, shSi, shSj);
  __syncthreads();
  att_star_mean_elu(shWh, 9, 64, shSi, shSj, shComb, 64);
  __syncthreads();

  // ===== Stage 5: lanes MLP (108 -> 64 relu -> 64 relu) -> comb[128:192] ======
  for (int idx = threadIdx.x; idx < 16 * 128; idx += 32) {
    int s = idx >> 7, k = idx & 127;
    float v = (k < 108) ? shCur[s * 192 + 56 + k] : 0.f;  // cur[120 + k]
    shX[idx] = (_Float16)v;
  }
  __syncthreads();
  {
    v8f acc[4] = {};
#pragma unroll
    for (int ks = 0; ks < 4; ++ks) {
      v16h a = ldA(shX, 128, ks * 32);
#pragma unroll
      for (int j = 0; j < 4; ++j)
        acc[j] = wmma_f16(a, ldB(wf + OFF_WL1, 64, ks * 32, 16 * j), acc[j]);
    }
#pragma unroll
    for (int j = 0; j < 4; ++j)
#pragma unroll
      for (int g = 0; g < 8; ++g) {
        float v = acc[j][g] + rb_l1[j];
        shHid[(cm + g) * 64 + 16 * j + cn] = (_Float16)(v > 0.f ? v : 0.f);
      }
    __syncthreads();
    v8f acc2[4] = {};
#pragma unroll
    for (int ks = 0; ks < 2; ++ks) {
      v16h a = ldA(shHid, 64, ks * 32);
#pragma unroll
      for (int j = 0; j < 4; ++j)
        acc2[j] = wmma_f16(a, ldB(wf + OFF_WL2, 64, ks * 32, 16 * j), acc2[j]);
    }
#pragma unroll
    for (int j = 0; j < 4; ++j)
#pragma unroll
      for (int g = 0; g < 8; ++g) {
        float v = acc2[j][g] + rb_l2[j];
        shComb[(cm + g) * 224 + 128 + 16 * j + cn] = (_Float16)(v > 0.f ? v : 0.f);
      }
  }
  __syncthreads();

  // ===== Stage 6: road + extra scalars -> comb[192:205], pad to 224 ==========
  for (int idx = threadIdx.x; idx < 16 * 32; idx += 32) {
    int s = idx >> 5, k = idx & 31;
    float v = 0.f;
    if (k < 13) {
      // cur cols {228,229,247,248,230,231,249,250,232,233,251,252,253} - 64
      const int mapr[13] = {164, 165, 183, 184, 166, 167, 185,
                            186, 168, 169, 187, 188, 189};
      v = shCur[s * 192 + mapr[k]];
    }
    shComb[s * 224 + 192 + k] = (_Float16)v;
  }
  __syncthreads();

  // ===== Stage 7: final MLP (205 -> 64 relu -> 64 relu) -> out ================
  {
    v8f acc[4] = {};
#pragma unroll
    for (int ks = 0; ks < 7; ++ks) {
      v16h a = ldA(shComb, 224, ks * 32);
#pragma unroll
      for (int j = 0; j < 4; ++j)
        acc[j] = wmma_f16(a, ldB(wf + OFF_WM1, 64, ks * 32, 16 * j), acc[j]);
    }
#pragma unroll
    for (int j = 0; j < 4; ++j)
#pragma unroll
      for (int g = 0; g < 8; ++g) {
        float v = acc[j][g] + rb_m1[j];
        shHid[(cm + g) * 64 + 16 * j + cn] = (_Float16)(v > 0.f ? v : 0.f);
      }
    __syncthreads();
    v8f acc2[4] = {};
#pragma unroll
    for (int ks = 0; ks < 2; ++ks) {
      v16h a = ldA(shHid, 64, ks * 32);
#pragma unroll
      for (int j = 0; j < 4; ++j)
        acc2[j] = wmma_f16(a, ldB(wf + OFF_WM2, 64, ks * 32, 16 * j), acc2[j]);
    }
#pragma unroll
    for (int j = 0; j < 4; ++j)
#pragma unroll
      for (int g = 0; g < 8; ++g) {
        float v = acc2[j][g] + rb_m2[j];
        out[(size_t)(s0 + cm + g) * 64 + 16 * j + cn] = v > 0.f ? v : 0.f;
      }
  }
}

extern "C" void kernel_launch(void* const* d_in, const int* in_sizes, int n_in,
                              void* d_out, int out_size, void* d_ws, size_t ws_size,
                              hipStream_t stream) {
  (void)in_sizes; (void)n_in; (void)out_size; (void)ws_size;
  _Float16* wf = (_Float16*)d_ws;

  auto cvt = [&](int idx, int K, int Kpad, int N, int off) {
    int tot = Kpad * N;
    cvt_pad_f16<<<(tot + 255) / 256, 256, 0, stream>>>(
        (const float*)d_in[idx], wf + off, K, Kpad, N);
  };
  cvt(1, 15, 32, 64, OFF_WSE);
  cvt(3, 64, 64, 32, OFF_WDE);
  cvt(5, 32, 32, 64, OFF_WH1);
  cvt(7, 32, 32, 64, OFF_WH2);
  cvt(9, 128, 128, 64, OFF_WHO);
  cvt(11, 6, 32, 16, OFF_WC1);
  cvt(13, 6, 32, 16, OFF_WC2);
  cvt(15, 32, 32, 64, OFF_WCO);
  cvt(17, 108, 128, 64, OFF_WL1);
  cvt(19, 64, 64, 64, OFF_WL2);
  cvt(21, 205, 224, 64, OFF_WM1);
  cvt(23, 64, 64, 64, OFF_WM2);

  pred_main<<<16384 / 16, 32, SMEM_BYTES, stream>>>(
      (const float*)d_in[0], (const float*)d_in[2], (const float*)d_in[4],
      (const float*)d_in[6], (const float*)d_in[8], (const float*)d_in[10],
      (const float*)d_in[12], (const float*)d_in[14], (const float*)d_in[16],
      (const float*)d_in[18], (const float*)d_in[20], (const float*)d_in[22],
      (const float*)d_in[24], wf, (float*)d_out);
}
